// HRViTAttention_31190052504216
// MI455X (gfx1250) — matrix operand order repaired and checked by reading, not compiled, not run on hardware
//
#include <hip/hip_runtime.h>

// ---------------------------------------------------------------------------
// HRViT cross-shaped attention block for MI455X (gfx1250, wave32, WMMA).
// All GEMM-class math runs on v_wmma_f32_16x16x32_f16 with fp32 accumulation;
// fp32 inputs are converted to f16 operands during LDS staging.  Pure-copy
// staging (f16 weights, Q/K rows) uses CDNA5 async global->LDS loads
// (global_load_async_to_lds_b128, ASYNCcnt) to bypass the VGPR round-trip.
//
// Workspace layout (assumed ws_size >= ~155 MB):
//   qv_h   : f16 [25088][768]   q (pre-scaled by dh^-0.5) and v
//   v_f32  : f32 [25088][384]   v for depthwise-conv branch
//   vconv  : f32 [25088][384]
//   attn   : f32 [25088][384]
//   WqkvT  : f16 [768][384]     (transposed, k-contiguous)
//   WoutT  : f16 [384][384]
//   sc/off : f32 [384] each     folded BatchNorm + bias
// ---------------------------------------------------------------------------

typedef _Float16 v8h  __attribute__((ext_vector_type(8)));
typedef _Float16 v16h __attribute__((ext_vector_type(16)));
typedef float    v8f  __attribute__((ext_vector_type(8)));

#define BATCH 8
#define HH    56
#define WW    56
#define NPIX  (HH * WW)          // 3136
#define MROWS (BATCH * NPIX)     // 25088
#define DIM   384
#define DH    48
#define LSEQ  392                // 7*56
#define LPAD  416                // padded to multiple of 32
#define DHP   64                 // dh padded to multiple of 32
#define SCALEQ 0.14433756729740643f  // 48^-0.5

#define BM 128
#define BN 128
#define BK 32

__device__ __forceinline__ v8f v8f_zero() {
  v8f z;
#pragma unroll
  for (int i = 0; i < 8; ++i) z[i] = 0.0f;
  return z;
}
__device__ __forceinline__ v8h v8h_zero() {
  v8h z;
#pragma unroll
  for (int i = 0; i < 8; ++i) z[i] = (_Float16)0.0f;
  return z;
}

__device__ __forceinline__ float hswish(float x) {
  return x * fminf(fmaxf(x + 3.0f, 0.0f), 6.0f) * (1.0f / 6.0f);
}

// CDNA5 async global->LDS copy, 16 bytes per lane, tracked by ASYNCcnt.
// VDST VGPR carries the LDS byte address (low 32 bits of the generic pointer).
__device__ __forceinline__ void async_copy16(void* lds_dst, const void* gsrc) {
  unsigned int l = (unsigned int)(unsigned long long)lds_dst;
  asm volatile("global_load_async_to_lds_b128 %0, %1, off"
               :
               : "v"(l), "v"(gsrc)
               : "memory");
}

__device__ __forceinline__ void wait_async0() {
#if __has_builtin(__builtin_amdgcn_s_wait_asynccnt)
  __builtin_amdgcn_s_wait_asynccnt(0);
#else
  asm volatile("s_wait_asynccnt 0x0" ::: "memory");
#endif
}

// A-matrix 16x32 f16 fragment (ISA 7.12.2): lane l<16 -> row l, K in {0..7,16..23};
// lane>=16 -> row l-16, K in {8..15,24..31}. base points at [row0][k0], k-contiguous.
__device__ __forceinline__ v16h load_a_frag(const _Float16* base, int row_stride, int lane) {
  const _Float16* p = base + (lane & 15) * row_stride + ((lane >> 4) << 3);
  v8h lo = *(const v8h*)(p);
  v8h hi = *(const v8h*)(p + 16);
  v16h a;
#pragma unroll
  for (int i = 0; i < 8; ++i) { a[i] = lo[i]; a[i + 8] = hi[i]; }
  return a;
}

// B-matrix 32x16 f16 fragment: lane l -> column l&15, K in {0..7,16..23} for l<16,
// {8..15,24..31} for l>=16 (matches A). Requires B stored [col][k] (k-contiguous).
__device__ __forceinline__ v16h load_b_frag(const _Float16* base, int col_stride, int lane) {
  const _Float16* p = base + (lane & 15) * col_stride + ((lane >> 4) << 3);
  v8h lo = *(const v8h*)(p);
  v8h hi = *(const v8h*)(p + 16);
  v16h b;
#pragma unroll
  for (int i = 0; i < 8; ++i) { b[i] = lo[i]; b[i + 8] = hi[i]; }
  return b;
}

__device__ __forceinline__ v8f wmma16(v16h a, v16h b, v8f c) {
  return __builtin_amdgcn_wmma_f32_16x16x32_f16(false, a, false, b, (short)0, c,
                                                false, false);
}

// ---------------------------------------------------------------------------
// K0: weight transpose + convert + BN folding
// ---------------------------------------------------------------------------
__global__ __launch_bounds__(256) void prep_kernel(
    const float* __restrict__ Wqkv, const float* __restrict__ Wout,
    const float* __restrict__ b_out, const float* __restrict__ gamma,
    const float* __restrict__ beta, const float* __restrict__ mean,
    const float* __restrict__ var, _Float16* __restrict__ WqkvT,
    _Float16* __restrict__ WoutT, float* __restrict__ sc,
    float* __restrict__ off) {
  int idx = blockIdx.x * 256 + threadIdx.x;
  if (idx < 768 * 384) {
    int n = idx / 384, k = idx % 384;
    WqkvT[idx] = (_Float16)Wqkv[k * 768 + n];
  }
  if (idx < 384 * 384) {
    int n = idx / 384, k = idx % 384;
    WootT:
    WoutT[idx] = (_Float16)Wout[k * 384 + n];
  }
  if (idx < 384) {
    float s = gamma[idx] * rsqrtf(var[idx] + 1e-5f);
    sc[idx] = s;
    off[idx] = beta[idx] + (b_out[idx] - mean[idx]) * s;
  }
}

// ---------------------------------------------------------------------------
// K1: qkv = x @ W_qkv + b.  M=25088, K=384, N=768.
// Writes q*SCALE and v as f16 (attention operands) and v as f32 (conv branch).
// ---------------------------------------------------------------------------
__global__ __launch_bounds__(256) void qkv_gemm(
    const float* __restrict__ x, const _Float16* __restrict__ WT,
    const float* __restrict__ bias, _Float16* __restrict__ qv_h,
    float* __restrict__ v_f32) {
  __shared__ alignas(64) _Float16 As[BM * BK];  // [row][k]
  __shared__ alignas(64) _Float16 Bs[BN * BK];  // [col][k]
  const int tid = threadIdx.x, lane = tid & 31, wave = tid >> 5;
  const int wm = wave & 3, wn = wave >> 2;
  const int row0 = blockIdx.y * BM, col0 = blockIdx.x * BN;

  v8f acc[2][4];
#pragma unroll
  for (int i = 0; i < 2; ++i)
#pragma unroll
    for (int j = 0; j < 4; ++j) acc[i][j] = v8f_zero();

  for (int k0 = 0; k0 < 384; k0 += BK) {
    {  // stage B via async global->LDS (pure f16 copy, no VGPR round-trip)
      int c = tid >> 1, kp = (tid & 1) * 16;
      const _Float16* src = WT + (size_t)(col0 + c) * 384 + k0 + kp;
      async_copy16(Bs + c * BK + kp, src);
      async_copy16(Bs + c * BK + kp + 8, src + 8);
    }
    {  // stage A (fp32 -> f16 conversion, must go through VALU)
      int r = tid >> 1, kp = (tid & 1) * 16;
      const float* src = x + (size_t)(row0 + r) * 384 + k0 + kp;
      _Float16* dst = As + r * BK + kp;
#pragma unroll
      for (int i = 0; i < 16; ++i) dst[i] = (_Float16)src[i];
      if (k0 + BK < 384) __builtin_prefetch(src + BK, 0, 1);
    }
    wait_async0();
    __syncthreads();
    v16h af[2], bf[4];
#pragma unroll
    for (int sm = 0; sm < 2; ++sm)
      af[sm] = load_a_frag(As + (wm * 32 + sm * 16) * BK, BK, lane);
#pragma unroll
    for (int sn = 0; sn < 4; ++sn)
      bf[sn] = load_b_frag(Bs + (wn * 64 + sn * 16) * BK, BK, lane);
#pragma unroll
    for (int sm = 0; sm < 2; ++sm)
#pragma unroll
      for (int sn = 0; sn < 4; ++sn) acc[sm][sn] = wmma16(af[sm], bf[sn], acc[sm][sn]);
    __syncthreads();
  }

#pragma unroll
  for (int sm = 0; sm < 2; ++sm)
#pragma unroll
    for (int sn = 0; sn < 4; ++sn) {
      int rbase = row0 + wm * 32 + sm * 16 + ((lane >> 4) << 3);
      int col = col0 + wn * 64 + sn * 16 + (lane & 15);
      float b = bias[col];
#pragma unroll
      for (int v = 0; v < 8; ++v) {
        int row = rbase + v;
        float val = acc[sm][sn][v] + b;
        if (col < 384) {
          qv_h[(size_t)row * 768 + col] = (_Float16)(val * SCALEQ);
        } else {
          qv_h[(size_t)row * 768 + col] = (_Float16)val;
          v_f32[(size_t)row * 384 + (col - 384)] = val;
        }
      }
    }
}

// ---------------------------------------------------------------------------
// K2: v_conv = dwconv3x3(hardswish(v)) + conv_b   (memory-bound, VALU)
// ---------------------------------------------------------------------------
__global__ __launch_bounds__(256) void dwconv_kernel(
    const float* __restrict__ v, const float* __restrict__ w,
    const float* __restrict__ bias, float* __restrict__ out) {
  int idx = blockIdx.x * 256 + threadIdx.x;
  if (idx >= MROWS * DIM) return;
  int c = idx % DIM;
  int bn = idx / DIM;
  int n = bn % NPIX, b = bn / NPIX;
  int y = n / WW, xx = n % WW;
  float acc = bias[c];
#pragma unroll
  for (int dy = 0; dy < 3; ++dy)
#pragma unroll
    for (int dx = 0; dx < 3; ++dx) {
      int yy = y + dy - 1, xc = xx + dx - 1;
      if (yy >= 0 && yy < HH && xc >= 0 && xc < WW) {
        float t = v[((size_t)b * NPIX + yy * WW + xc) * DIM + c];
        acc += hswish(t) * w[c * 9 + dy * 3 + dx];
      }
    }
  out[idx] = acc;
}

// ---------------------------------------------------------------------------
// K3: striped cross attention.  One block per (dir, b, stripe g, head h).
// blockDim=128 (4 waves).  Dynamic LDS = 306176 B (uses CDNA5 320KB/WGP).
// ---------------------------------------------------------------------------
__global__ __launch_bounds__(128) void attn_kernel(
    const _Float16* __restrict__ qv_h, float* __restrict__ attn_out) {
  extern __shared__ char smem[];
  _Float16* Klds = (_Float16*)smem;                 // [416][64]  keys, row-major
  _Float16* Qlds = Klds + LPAD * DHP;               // [416][64]  queries
  _Float16* Vlds = Qlds + LPAD * DHP;               // [48][416]  values, transposed
  float*    Slds = (float*)(Vlds + DH * LPAD);      // 4 * [16][416] fp32 scores
  _Float16* Plds = (_Float16*)(Slds + 4 * 16 * LPAD);  // 4 * [16][416] f16 probs

  const int bid = blockIdx.x;
  const int h = bid & 3, g = (bid >> 2) & 7, b = (bid >> 5) & 7, dir = bid >> 8;
  const int tid = threadIdx.x, lane = tid & 31, wave = tid >> 5;

  const int qch = dir * 192 + h * DH;        // q channels in [0,384)
  const int kch = 384 + dir * 192 + h * DH;  // k==v channels

  // ---- phase A: async-copy valid Q/K rows; zero-fill padding ----
  v8h z = v8h_zero();
  for (int m = tid; m < LPAD; m += 128) {
    if (m < LSEQ) {
      int n = (dir == 0) ? (g * LSEQ + m) : ((m % 56) * 56 + g * 7 + (m / 56));
      const _Float16* row = qv_h + ((size_t)b * NPIX + (size_t)n) * 768;
#pragma unroll
      for (int cc = 0; cc < 6; ++cc) {
        async_copy16(Klds + m * DHP + cc * 8, row + kch + cc * 8);
        async_copy16(Qlds + m * DHP + cc * 8, row + qch + cc * 8);
      }
      *(v8h*)(Klds + m * DHP + 48) = z;
      *(v8h*)(Klds + m * DHP + 56) = z;
      *(v8h*)(Qlds + m * DHP + 48) = z;
      *(v8h*)(Qlds + m * DHP + 56) = z;
    } else {
#pragma unroll
      for (int cc = 0; cc < 8; ++cc) {
        *(v8h*)(Klds + m * DHP + cc * 8) = z;
        *(v8h*)(Qlds + m * DHP + cc * 8) = z;
      }
    }
  }
  wait_async0();
  __syncthreads();

  // ---- phase B: build transposed V from staged K (LDS->LDS) ----
  for (int m = tid; m < LPAD; m += 128) {
#pragma unroll
    for (int cc = 0; cc < 6; ++cc) {
      v8h kv = *(const v8h*)(Klds + m * DHP + cc * 8);
#pragma unroll
      for (int i = 0; i < 8; ++i) Vlds[(cc * 8 + i) * LPAD + m] = kv[i];
    }
  }
  __syncthreads();

  float* Sw = Slds + wave * 16 * LPAD;
  _Float16* Pw = Plds + wave * 16 * LPAD;

  for (int rb = wave; rb < 25; rb += 4) {
    const int mrow0 = rb * 16;
    // ---- S = (scaled Q) K^T : 26 key tiles x 2 k-steps of 32 ----
    v16h aq0 = load_a_frag(Qlds + mrow0 * DHP + 0, DHP, lane);
    v16h aq1 = load_a_frag(Qlds + mrow0 * DHP + 32, DHP, lane);
    for (int j = 0; j < 26; ++j) {
      v8f c = v8f_zero();
      v16h b0 = load_b_frag(Klds + (j * 16) * DHP + 0, DHP, lane);
      v16h b1 = load_b_frag(Klds + (j * 16) * DHP + 32, DHP, lane);
      c = wmma16(aq0, b0, c);
      c = wmma16(aq1, b1, c);
      int ccol = j * 16 + (lane & 15);
      int rrow = (lane >> 4) << 3;
#pragma unroll
      for (int v = 0; v < 8; ++v) Sw[(rrow + v) * LPAD + ccol] = c[v];
    }
    // ---- row softmax over 392 valid keys ----
    for (int r = 0; r < 16; ++r) {
      float* srow = Sw + r * LPAD;
      float mx = -3.0e38f;
      for (int c = lane; c < LSEQ; c += 32) mx = fmaxf(mx, srow[c]);
#pragma unroll
      for (int o = 16; o > 0; o >>= 1) mx = fmaxf(mx, __shfl_xor(mx, o, 32));
      float sum = 0.0f;
      for (int c = lane; c < LSEQ; c += 32) {
        float e = __expf(srow[c] - mx);
        srow[c] = e;
        sum += e;
      }
#pragma unroll
      for (int o = 16; o > 0; o >>= 1) sum += __shfl_xor(sum, o, 32);
      float inv = 1.0f / sum;
      _Float16* prow = Pw + r * LPAD;
      for (int c = lane; c < LPAD; c += 32)
        prow[c] = (c < LSEQ) ? (_Float16)(srow[c] * inv) : (_Float16)0.0f;
    }
    // ---- O = P V : 3 channel tiles x 13 k-steps of 32 ----
#pragma unroll
    for (int t = 0; t < 3; ++t) {
      v8f c = v8f_zero();
      for (int ks = 0; ks < 13; ++ks) {
        v16h a = load_a_frag(Pw + ks * 32, LPAD, lane);
        v16h bv = load_b_frag(Vlds + (t * 16) * LPAD + ks * 32, LPAD, lane);
        c = wmma16(a, bv, c);
      }
      int ch = dir * 192 + h * DH + t * 16 + (lane & 15);
      int rrow = (lane >> 4) << 3;
#pragma unroll
      for (int v = 0; v < 8; ++v) {
        int m = mrow0 + rrow + v;
        if (m < LSEQ) {
          int n = (dir == 0) ? (g * LSEQ + m) : ((m % 56) * 56 + g * 7 + (m / 56));
          attn_out[((size_t)b * NPIX + n) * DIM + ch] = c[v];
        }
      }
    }
  }
}

// ---------------------------------------------------------------------------
// K4: out = hardswish(attn + vconv) @ W_out, fused bias + BatchNorm epilogue.
// ---------------------------------------------------------------------------
__global__ __launch_bounds__(256) void out_gemm(
    const float* __restrict__ attn, const float* __restrict__ vconv,
    const _Float16* __restrict__ WT, const float* __restrict__ sc,
    const float* __restrict__ off, float* __restrict__ out) {
  __shared__ alignas(64) _Float16 As[BM * BK];
  __shared__ alignas(64) _Float16 Bs[BN * BK];
  const int tid = threadIdx.x, lane = tid & 31, wave = tid >> 5;
  const int wm = wave & 3, wn = wave >> 2;
  const int row0 = blockIdx.y * BM, col0 = blockIdx.x * BN;

  v8f acc[2][4];
#pragma unroll
  for (int i = 0; i < 2; ++i)
#pragma unroll
    for (int j = 0; j < 4; ++j) acc[i][j] = v8f_zero();

  for (int k0 = 0; k0 < 384; k0 += BK) {
    {  // stage B via async global->LDS
      int c = tid >> 1, kp = (tid & 1) * 16;
      const _Float16* src = WT + (size_t)(col0 + c) * 384 + k0 + kp;
      async_copy16(Bs + c * BK + kp, src);
      async_copy16(Bs + c * BK + kp + 8, src + 8);
    }
    {  // stage A: hardswish(attn + vconv) -> f16
      int r = tid >> 1, kp = (tid & 1) * 16;
      size_t base = (size_t)(row0 + r) * 384 + k0 + kp;
      _Float16* dst = As + r * BK + kp;
#pragma unroll
      for (int i = 0; i < 16; ++i)
        dst[i] = (_Float16)hswish(attn[base + i] + vconv[base + i]);
    }
    wait_async0();
    __syncthreads();
    v16h af[2], bf[4];
#pragma unroll
    for (int sm = 0; sm < 2; ++sm)
      af[sm] = load_a_frag(As + (wm * 32 + sm * 16) * BK, BK, lane);
#pragma unroll
    for (int sn = 0; sn < 4; ++sn)
      bf[sn] = load_b_frag(Bs + (wn * 64 + sn * 16) * BK, BK, lane);
#pragma unroll
    for (int sm = 0; sm < 2; ++sm)
#pragma unroll
      for (int sn = 0; sn < 4; ++sn) acc[sm][sn] = wmma16(af[sm], bf[sn], acc[sm][sn]);
    __syncthreads();
  }

#pragma unroll
  for (int sm = 0; sm < 2; ++sm)
#pragma unroll
    for (int sn = 0; sn < 4; ++sn) {
      int rbase = row0 + wm * 32 + sm * 16 + ((lane >> 4) << 3);
      int col = col0 + wn * 64 + sn * 16 + (lane & 15);
      float s = sc[col], o = off[col];
#pragma unroll
      for (int v = 0; v < 8; ++v) {
        int row = rbase + v;
        out[(size_t)row * 384 + col] = acc[sm][sn][v] * s + o;
      }
    }
}

// ---------------------------------------------------------------------------
extern "C" void kernel_launch(void* const* d_in, const int* in_sizes, int n_in,
                              void* d_out, int out_size, void* d_ws,
                              size_t ws_size, hipStream_t stream) {
  (void)in_sizes; (void)n_in; (void)out_size; (void)ws_size;
  const float* x       = (const float*)d_in[0];
  const float* W_qkv   = (const float*)d_in[1];
  const float* b_qkv   = (const float*)d_in[2];
  const float* conv_w  = (const float*)d_in[3];
  const float* conv_b  = (const float*)d_in[4];
  const float* W_out   = (const float*)d_in[5];
  const float* b_out   = (const float*)d_in[6];
  const float* bn_g    = (const float*)d_in[7];
  const float* bn_b    = (const float*)d_in[8];
  const float* bn_m    = (const float*)d_in[9];
  const float* bn_v    = (const float*)d_in[10];
  float* out = (float*)d_out;

  auto aup = [](size_t v) { return (v + 255) & ~(size_t)255; };
  char* ws = (char*)d_ws;
  size_t o = 0;
  _Float16* qv_h  = (_Float16*)(ws + o); o += aup((size_t)MROWS * 768 * 2);
  float*    v_f32 = (float*)(ws + o);    o += aup((size_t)MROWS * 384 * 4);
  float*    vconv = (float*)(ws + o);    o += aup((size_t)MROWS * 384 * 4);
  float*    attn  = (float*)(ws + o);    o += aup((size_t)MROWS * 384 * 4);
  _Float16* WqkvT = (_Float16*)(ws + o); o += aup((size_t)768 * 384 * 2);
  _Float16* WoutT = (_Float16*)(ws + o); o += aup((size_t)384 * 384 * 2);
  float*    sc    = (float*)(ws + o);    o += aup(384 * 4);
  float*    off   = (float*)(ws + o);    o += aup(384 * 4);

  prep_kernel<<<(768 * 384 + 255) / 256, 256, 0, stream>>>(
      W_qkv, W_out, b_out, bn_g, bn_b, bn_m, bn_v, WqkvT, WoutT, sc, off);

  qkv_gemm<<<dim3(768 / BN, MROWS / BM), 256, 0, stream>>>(x, WqkvT, b_qkv,
                                                           qv_h, v_f32);

  dwconv_kernel<<<(MROWS * DIM + 255) / 256, 256, 0, stream>>>(v_f32, conv_w,
                                                               conv_b, vconv);

  const size_t attn_lds =
      (size_t)LPAD * DHP * 2 * 2 +            // K + Q
      (size_t)DH * LPAD * 2 +                 // V transposed
      (size_t)4 * 16 * LPAD * 4 +             // S fp32 (per wave)
      (size_t)4 * 16 * LPAD * 2;              // P f16 (per wave)   = 306176 B
  attn_kernel<<<512, 128, attn_lds, stream>>>(qv_h, attn);

  out_gemm<<<dim3(384 / BN, MROWS / BM), 256, 0, stream>>>(attn, vconv, WoutT,
                                                           sc, off, out);
}
